// GPT2_42382737277040
// MI455X (gfx1250) — matrix operand order repaired and checked by reading, not compiled, not run on hardware
//
#include <hip/hip_runtime.h>
#include <hip/hip_bf16.h>

// ---------------- types / constants ----------------
typedef __attribute__((ext_vector_type(16))) __bf16 v16bf;
typedef __attribute__((ext_vector_type(8)))  float  v8f;

#define WMMA_BF16(A, B, C) \
  __builtin_amdgcn_wmma_f32_16x16x32_bf16(false, (A), false, (B), (short)0, (C), false, false)

static constexpr int Lc  = 12;
static constexpr int Hc  = 12;
static constexpr int Ec  = 768;
static constexpr int Vc  = 50257;
static constexpr int Bc  = 4;
static constexpr int Sc_ = 1024;
static constexpr int Mc  = Bc * Sc_;     // 4096 rows
static constexpr int E3c = 3 * Ec;       // 2304
static constexpr int E4c = 4 * Ec;       // 3072
static constexpr int Dc  = Ec / Hc;      // 64 head dim

// ---------------- helpers ----------------
__device__ inline float gelu_tanh(float x) {
  const float c = 0.7978845608028654f;
  float t = tanhf(c * (x + 0.044715f * x * x * x));
  return 0.5f * x * (1.0f + t);
}

// Convert W[K,N] fp32 -> Wt[N,K] bf16 (transposed so GEMM B-fragments are
// contiguous along K per lane, matching the 16x16x32 B layout).
__global__ void k_convT(const float* __restrict__ W, __bf16* __restrict__ Wt,
                        int K, int N) {
  long long id = (long long)blockIdx.x * 256 + threadIdx.x;
  long long total = (long long)K * N;
  if (id >= total) return;
  int n = (int)(id / K);
  int k = (int)(id % K);
  Wt[id] = (__bf16)W[(long long)k * N + n];
}

// h[row,:] = wte[x[row]] + wpe[row % S]
__global__ void k_embed(const int* __restrict__ x, const float* __restrict__ wte,
                        const float* __restrict__ wpe, float* __restrict__ h) {
  int row = blockIdx.x;
  int s = row & (Sc_ - 1);
  int tok = x[row];
  for (int e = threadIdx.x; e < Ec; e += 256)
    h[(long long)row * Ec + e] = wte[(long long)tok * Ec + e] + wpe[(long long)s * Ec + e];
}

// LayerNorm over E=768, writes bf16 (GEMM A input)
__global__ __launch_bounds__(256) void k_ln(const float* __restrict__ h,
                                            const float* __restrict__ g,
                                            const float* __restrict__ bta,
                                            __bf16* __restrict__ out) {
  __shared__ float r1[256], r2[256];
  int row = blockIdx.x, tid = threadIdx.x;
  float s = 0.f, s2 = 0.f;
  for (int e = tid; e < Ec; e += 256) {
    float v = h[(long long)row * Ec + e];
    s += v; s2 += v * v;
  }
  r1[tid] = s; r2[tid] = s2;
  __syncthreads();
  for (int st = 128; st > 0; st >>= 1) {
    if (tid < st) { r1[tid] += r1[tid + st]; r2[tid] += r2[tid + st]; }
    __syncthreads();
  }
  float mean = r1[0] * (1.0f / Ec);
  float var  = r2[0] * (1.0f / Ec) - mean * mean;
  float rstd = rsqrtf(var + 1e-5f);
  for (int e = tid; e < Ec; e += 256) {
    float v = (h[(long long)row * Ec + e] - mean) * rstd * g[e] + bta[e];
    out[(long long)row * Ec + e] = (__bf16)v;
  }
}

// Generic bf16 WMMA GEMM: C[M,N] = A[M,K] @ Wt[N,K]^T (+bias)(+resid)(gelu?)
// 256 threads = 8 wave32s; workgroup tile 128x128; K-step 32.
// Wave w owns a 64x32 sub-tile: 4 m-tiles x 2 n-tiles -> 8 WMMA per K-step,
// A-fragments reused across n-tiles, B-fragments across m-tiles (1.5 DS
// loads per WMMA instead of 6).
__global__ __launch_bounds__(256) void k_gemm(const __bf16* __restrict__ A,
                                              const __bf16* __restrict__ Wt,
                                              const float* __restrict__ bias,
                                              const float* __restrict__ resid,
                                              float* __restrict__ outF,
                                              __bf16* __restrict__ outB,
                                              int N, int K, int do_gelu) {
  __shared__ __align__(32) __bf16 As[128 * 32];   // A tile [m][k]
  __shared__ __align__(32) __bf16 Bts[128 * 32];  // W tile [n][k] (pre-transposed)

  int tid = threadIdx.x;
  int w = tid >> 5, lane = tid & 31, lo = lane & 15, hi = lane >> 4;
  int wm = (w & 1) * 64;   // wave's m offset within WG tile (0 or 64)
  int wn = (w >> 1) * 32;  // wave's n offset within WG tile (0,32,64,96)
  int m0 = blockIdx.y * 128;
  int n0 = blockIdx.x * 128;

  v8f acc[4][2];
#pragma unroll
  for (int i = 0; i < 4; ++i)
#pragma unroll
    for (int j = 0; j < 2; ++j)
      acc[i][j] = (v8f){0.f,0.f,0.f,0.f,0.f,0.f,0.f,0.f};

  int srow = tid >> 1;          // 0..127 staging row
  int sseg = (tid & 1) * 16;    // 0 or 16 (two int4 per thread)
  int gn = n0 + srow;

  for (int k0 = 0; k0 < K; k0 += 32) {
    // cooperative stage: each thread copies 16 bf16 of A and of W
    const __bf16* ag = &A[(long long)(m0 + srow) * K + k0 + sseg];
    *(int4*)&As[srow * 32 + sseg]     = *(const int4*)ag;
    *(int4*)&As[srow * 32 + sseg + 8] = *(const int4*)(ag + 8);
    int4 z; z.x = z.y = z.z = z.w = 0;
    int4 b0v = z, b1v = z;
    const __bf16* wg = &Wt[(long long)gn * K + k0 + sseg];
    if (gn < N) { b0v = *(const int4*)wg; b1v = *(const int4*)(wg + 8); }
    *(int4*)&Bts[srow * 32 + sseg]     = b0v;
    *(int4*)&Bts[srow * 32 + sseg + 8] = b1v;
    // prefetch next K-step tiles into cache (global_prefetch_b8)
    if (k0 + 32 < K) {
      __builtin_prefetch(ag + 32, 0, 0);
      if (gn < N) __builtin_prefetch(wg + 32, 0, 0);
    }
    __syncthreads();

    // A fragments: a[e]: m = lo, k = hi*16 + e
    union { unsigned u[8]; v16bf v; } af[4];
#pragma unroll
    for (int i = 0; i < 4; ++i) {
      int am = wm + i * 16 + lo;
#pragma unroll
      for (int r = 0; r < 8; ++r) {
        int kb = hi * 16 + r * 2;
        af[i].u[r] = *(const unsigned*)&As[am * 32 + kb];
      }
    }
    // B fragments: n = lo (within tile), k = e + 16*hi -> contiguous row
    v16bf bf0 = *(const v16bf*)&Bts[(wn + lo) * 32 + hi * 16];
    v16bf bf1 = *(const v16bf*)&Bts[(wn + 16 + lo) * 32 + hi * 16];

#pragma unroll
    for (int i = 0; i < 4; ++i) {
      acc[i][0] = WMMA_BF16(af[i].v, bf0, acc[i][0]);
      acc[i][1] = WMMA_BF16(af[i].v, bf1, acc[i][1]);
    }
    __syncthreads();
  }

  // epilogue
#pragma unroll
  for (int i = 0; i < 4; ++i) {
#pragma unroll
    for (int j = 0; j < 2; ++j) {
#pragma unroll
      for (int r = 0; r < 8; ++r) {
        int m = m0 + wm + i * 16 + r + hi * 8;
        int n = n0 + wn + j * 16 + lo;
        if (n < N) {
          float v = acc[i][j][r];
          if (bias)  v += bias[n];
          if (resid) v += resid[(long long)m * N + n];
          if (do_gelu) v = gelu_tanh(v);
          if (outF) outF[(long long)m * N + n] = v;
          if (outB) outB[(long long)m * N + n] = (__bf16)v;
        }
      }
    }
  }
}

// Causal attention, one workgroup (4 wave32s) per (b, head, 16-query tile).
// qkv: bf16 [B*S, 3E]; y: bf16 [B*S, E]. Scores and P live in LDS.
static constexpr int SMEM_SC   = 16 * 1024 * 4;   // scores f32
static constexpr int SMEM_PB   = 16 * 1024 * 2;   // P bf16 (unnormalized exp)
static constexpr int SMEM_QS   = 16 * 64 * 2;     // Q tile bf16
static constexpr int SMEM_VT   = 64 * 32 * 2;     // V chunk transposed bf16
static constexpr int SMEM_RED  = 16 * 8 * 4;
static constexpr int SMEM_ROW  = 16 * 4;
static constexpr int ATTN_SMEM = SMEM_SC + SMEM_PB + SMEM_QS + SMEM_VT + SMEM_RED + 2 * SMEM_ROW;

__global__ __launch_bounds__(128) void k_attn(const __bf16* __restrict__ qkv,
                                              __bf16* __restrict__ y) {
  extern __shared__ char smem[];
  float*  Sco   = (float*)smem;                                  // [16][1024]
  __bf16* Pb    = (__bf16*)(smem + SMEM_SC);                     // [16][1024]
  __bf16* Qs    = (__bf16*)(smem + SMEM_SC + SMEM_PB);           // [16][64]
  __bf16* Vt    = (__bf16*)(smem + SMEM_SC + SMEM_PB + SMEM_QS); // [64][32]
  float*  red   = (float*)(smem + SMEM_SC + SMEM_PB + SMEM_QS + SMEM_VT);
  float*  rowmx = red + 16 * 8;
  float*  rowin = rowmx + 16;

  int qt = blockIdx.x, hh = blockIdx.y, bb = blockIdx.z;
  int tid = threadIdx.x, w = tid >> 5, lane = tid & 31, lo = lane & 15, hi = lane >> 4;
  int q0 = qt * 16;

  // stage Q tile (bf16, contiguous per row)
  {
    int row = tid >> 3, c = (tid & 7) * 8;
    *(int4*)&Qs[row * 64 + c] =
        *(const int4*)&qkv[(long long)(bb * Sc_ + q0 + row) * E3c + hh * Dc + c];
  }
  __syncthreads();

  // Q fragments (d 0..31 and 32..63)
  union { unsigned u[8]; v16bf v; } qa0, qa1;
#pragma unroll
  for (int r = 0; r < 8; ++r) {
    int kb = hi * 16 + r * 2;
    qa0.u[r] = *(const unsigned*)&Qs[lo * 64 + kb];
    qa1.u[r] = *(const unsigned*)&Qs[lo * 64 + 32 + kb];
  }

  // ---- scores: S[16, (qt+1)*16] = q @ k^T * 1/8, causal-masked ----
  for (int kt = w; kt <= qt; kt += 4) {
    const __bf16* kb = qkv + (long long)(bb * Sc_ + kt * 16 + lo) * E3c
                           + Ec + hh * Dc + hi * 16;
    v16bf kf0 = *(const v16bf*)kb;          // d = hi*16 + e (0..31)
    v16bf kf1 = *(const v16bf*)(kb + 32);   // d 32..63
    v8f s = {0.f,0.f,0.f,0.f,0.f,0.f,0.f,0.f};
    s = WMMA_BF16(qa0.v, kf0, s);
    s = WMMA_BF16(qa1.v, kf1, s);
#pragma unroll
    for (int r = 0; r < 8; ++r) {
      int m = r + hi * 8;
      float v = s[r] * 0.125f;                      // 1/sqrt(64)
      if (kt == qt && lo > m) v = -1e30f;           // causal mask on diagonal
      Sco[m * 1024 + kt * 16 + lo] = v;
    }
  }
  __syncthreads();

  // ---- softmax in LDS: P = exp(s - rowmax) (bf16), rowin = 1/rowsum ----
  int ncols = (qt + 1) * 16;
  {
    int row = tid & 15, sub = tid >> 4;  // 8 partials per row
    float mx = -1e30f;
    for (int j = sub; j < ncols; j += 8) mx = fmaxf(mx, Sco[row * 1024 + j]);
    red[row * 8 + sub] = mx;
    __syncthreads();
    if (sub == 0) {
      float m2 = red[row * 8];
      for (int i = 1; i < 8; ++i) m2 = fmaxf(m2, red[row * 8 + i]);
      rowmx[row] = m2;
    }
    __syncthreads();
    float rmax = rowmx[row];
    float sm = 0.f;
    for (int j = sub; j < ncols; j += 8) {
      float p = __expf(Sco[row * 1024 + j] - rmax);
      Pb[row * 1024 + j] = (__bf16)p;
      sm += p;
    }
    red[row * 8 + sub] = sm;
    __syncthreads();
    if (sub == 0) {
      float s2 = 0.f;
      for (int i = 0; i < 8; ++i) s2 += red[row * 8 + i];
      rowin[row] = 1.0f / s2;
    }
    // zero-pad P to a 32-key boundary for the K=32 WMMA chunks
    if ((ncols & 31) != 0) {
      for (int idx = tid; idx < 256; idx += 128)
        Pb[(idx >> 4) * 1024 + ncols + (idx & 15)] = (__bf16)0.0f;
    }
  }
  __syncthreads();

  // ---- y = P @ V, wave w owns d-tile [w*16, w*16+16) ----
  int nch = (ncols + 31) >> 5;
  v8f yacc = {0.f,0.f,0.f,0.f,0.f,0.f,0.f,0.f};
  for (int kc = 0; kc < nch; ++kc) {
    // stage V chunk transposed: Vt[d][key_local]
    {
      int key_l = tid & 31, dh = tid >> 5;
      const __bf16* vb = qkv + (long long)(bb * Sc_ + kc * 32 + key_l) * E3c
                             + 2 * Ec + hh * Dc + dh * 16;
      union { int4 q[2]; __bf16 h[16]; } vv;
      vv.q[0] = *(const int4*)vb;
      vv.q[1] = *(const int4*)(vb + 8);
#pragma unroll
      for (int i = 0; i < 16; ++i) Vt[(dh * 16 + i) * 32 + key_l] = vv.h[i];
    }
    __syncthreads();

    union { unsigned u[8]; v16bf v; } pf;
#pragma unroll
    for (int r = 0; r < 8; ++r) {
      int kb = hi * 16 + r * 2;
      pf.u[r] = *(const unsigned*)&Pb[lo * 1024 + kc * 32 + kb];
    }
    v16bf vf = *(const v16bf*)&Vt[(w * 16 + lo) * 32 + hi * 16];
    yacc = WMMA_BF16(pf.v, vf, yacc);
    __syncthreads();
  }

  // store y (apply 1/rowsum here; softmax is linear in the scale)
#pragma unroll
  for (int r = 0; r < 8; ++r) {
    int m = r + hi * 8;
    float val = yacc[r] * rowin[m];
    y[(long long)(bb * Sc_ + q0 + m) * Ec + hh * Dc + w * 16 + lo] = (__bf16)val;
  }
}

// ---------------- orchestration ----------------
extern "C" void kernel_launch(void* const* d_in, const int* in_sizes, int n_in,
                              void* d_out, int out_size, void* d_ws, size_t ws_size,
                              hipStream_t stream) {
  (void)in_sizes; (void)n_in; (void)out_size; (void)ws_size;
  const int*   x         = (const int*)  d_in[0];
  const float* wte       = (const float*)d_in[1];
  const float* wpe       = (const float*)d_in[2];
  const float* ln1_g     = (const float*)d_in[3];
  const float* ln1_b     = (const float*)d_in[4];
  const float* attn_w    = (const float*)d_in[5];
  const float* attn_b    = (const float*)d_in[6];
  const float* attn_pw   = (const float*)d_in[7];
  const float* attn_pb   = (const float*)d_in[8];
  const float* ln2_g     = (const float*)d_in[9];
  const float* ln2_b     = (const float*)d_in[10];
  const float* fc_w      = (const float*)d_in[11];
  const float* fc_b      = (const float*)d_in[12];
  const float* fc_pw     = (const float*)d_in[13];
  const float* fc_pb     = (const float*)d_in[14];
  const float* lnf_g     = (const float*)d_in[15];
  const float* lnf_b     = (const float*)d_in[16];
  const float* lm_head_w = (const float*)d_in[17];

  // carve workspace
  char* p = (char*)d_ws;
  auto carve = [&](size_t bytes) {
    char* r = p;
    p += (bytes + 255) & ~(size_t)255;
    return r;
  };
  float*  h    = (float*) carve((size_t)Mc * Ec * 4);    // residual stream
  __bf16* bfA  = (__bf16*)carve((size_t)Mc * Ec * 2);    // LN outputs
  __bf16* mlp  = (__bf16*)carve((size_t)Mc * E4c * 2);   // gelu(fc) bf16
  __bf16* qkvb = (__bf16*)carve((size_t)Mc * E3c * 2);   // qkv bf16
  __bf16* ybf  = (__bf16*)carve((size_t)Mc * Ec * 2);    // attention out bf16
  __bf16* Wt   = (__bf16*)carve((size_t)Vc * Ec * 2);    // transposed bf16 weights

  auto cblocks = [](long long total) { return (int)((total + 255) / 256); };
  auto ggrid = [](int N) { return dim3((unsigned)((N + 127) / 128), (unsigned)(Mc / 128)); };

  k_embed<<<Mc, 256, 0, stream>>>(x, wte, wpe, h);

  for (int l = 0; l < Lc; ++l) {
    // ln1 -> qkv GEMM -> attention
    k_ln<<<Mc, 256, 0, stream>>>(h, ln1_g + l * Ec, ln1_b + l * Ec, bfA);
    k_convT<<<cblocks((long long)Ec * E3c), 256, 0, stream>>>(
        attn_w + (long long)l * Ec * E3c, Wt, Ec, E3c);
    k_gemm<<<ggrid(E3c), 256, 0, stream>>>(
        bfA, Wt, attn_b + l * E3c, nullptr, nullptr, qkvb, E3c, Ec, 0);
    k_attn<<<dim3(Sc_ / 16, Hc, Bc), 128, ATTN_SMEM, stream>>>(qkvb, ybf);

    // attention projection (+residual into h)
    k_convT<<<cblocks((long long)Ec * Ec), 256, 0, stream>>>(
        attn_pw + (long long)l * Ec * Ec, Wt, Ec, Ec);
    k_gemm<<<ggrid(Ec), 256, 0, stream>>>(
        ybf, Wt, attn_pb + l * Ec, h, h, nullptr, Ec, Ec, 0);

    // ln2 -> fc (gelu) -> fc_proj (+residual into h)
    k_ln<<<Mc, 256, 0, stream>>>(h, ln2_g + l * Ec, ln2_b + l * Ec, bfA);
    k_convT<<<cblocks((long long)Ec * E4c), 256, 0, stream>>>(
        fc_w + (long long)l * Ec * E4c, Wt, Ec, E4c);
    k_gemm<<<ggrid(E4c), 256, 0, stream>>>(
        bfA, Wt, fc_b + l * E4c, nullptr, nullptr, mlp, E4c, Ec, 1);
    k_convT<<<cblocks((long long)E4c * Ec), 256, 0, stream>>>(
        fc_pw + (long long)l * E4c * Ec, Wt, E4c, Ec);
    k_gemm<<<ggrid(Ec), 256, 0, stream>>>(
        mlp, Wt, fc_pb + l * Ec, h, h, nullptr, Ec, E4c, 0);
  }

  // final LN + lm_head -> logits (f32 out, N=50257 padded with guards)
  k_ln<<<Mc, 256, 0, stream>>>(h, lnf_g, lnf_b, bfA);
  k_convT<<<cblocks((long long)Ec * Vc), 256, 0, stream>>>(lm_head_w, Wt, Ec, Vc);
  k_gemm<<<ggrid(Vc), 256, 0, stream>>>(
      bfA, Wt, nullptr, nullptr, (float*)d_out, nullptr, Vc, Ec, 0);
}
